// GRU_87007447482534
// MI455X (gfx1250) — matrix operand order repaired and checked by reading, not compiled, not run on hardware
//
#include <hip/hip_runtime.h>

// ---------------------------------------------------------------------------
// Problem constants (match the reference)
// ---------------------------------------------------------------------------
#define L_LAYERS 8
#define H        32
#define IN_DIM   4
#define B_BATCH  1024
#define S_SEQ    512
#define ODE_N    40
#define EPS_BN   1e-5f

typedef __attribute__((ext_vector_type(16))) _Float16 v16h;
typedef __attribute__((ext_vector_type(8)))  _Float16 v8h;
typedef __attribute__((ext_vector_type(2)))  _Float16 v2h;
typedef __attribute__((ext_vector_type(8)))  float    v8f;
typedef __attribute__((ext_vector_type(2)))  float    v2f;

__device__ __forceinline__ float sigmoidf_(float x) {
    // rcp(1 + exp(-x)) : v_mul + v_exp + v_add + v_rcp
    return __builtin_amdgcn_rcpf(1.0f + __expf(-x));
}
__device__ __forceinline__ v8f splat8(float x) {
    v8f v;
#pragma unroll
    for (int i = 0; i < 8; ++i) v[i] = x;
    return v;
}

// U fragment LDS layout: per (gate, nt): 32 lanes x 24 halfs (16 used, 48B
// stride => 16B aligned, 2-way bank conflict max).
#define UFRAG_STRIDE 24

// ---------------------------------------------------------------------------
// Persistent GRU kernel: ONE workgroup (32 wave32s) on one WGP.
//   h lives in LDS (f32 master). Per step:
//     phase 1: WMMA  h@U (f16 16x16x32, K=H) + x@W (f32 16x16x4, K=IN) + bias
//     phase 2: batch-global BatchNorm stats (butterfly + LDS partials,
//              recomputed redundantly per wave -> only 2 barriers/step)
//     phase 3: normalize, relu, GRU update, write h back to LDS
//   Last layer additionally projects hn @ Wo + bo -> outs[B,S,4] (workspace).
// B-operand fragments are wave-invariant => staged in LDS per layer, loaded
// per use. z is held packed f16 in registers to stay under the VGPR budget
// (no scratch spills in the 4096-step serial loop).
// ---------------------------------------------------------------------------
__global__ __launch_bounds__(1024, 1)
void gru_persistent(const float* __restrict__ in_seq,
                    const float* __restrict__ Wr_w, const float* __restrict__ Wr_b,
                    const float* __restrict__ Ur_w, const float* __restrict__ Ur_b,
                    const float* __restrict__ Wz_w, const float* __restrict__ Wz_b,
                    const float* __restrict__ Uz_w, const float* __restrict__ Uz_b,
                    const float* __restrict__ Wh_w, const float* __restrict__ Wh_b,
                    const float* __restrict__ Uh_w, const float* __restrict__ Uh_b,
                    const float* __restrict__ out_w, const float* __restrict__ out_b,
                    const float* __restrict__ bn_g, const float* __restrict__ bn_b,
                    float* __restrict__ outs)       // [B, S, 4] workspace
{
    __shared__ __align__(16) float    h_lds[B_BATCH * H];             // 128 KB
    __shared__ __align__(16) _Float16 ufrag[3][2][32 * UFRAG_STRIDE]; // ~9 KB
    __shared__ __align__(16) float    wfrag[3][2][32 * 2];            // 1.5 KB
    __shared__ float red_s[32];
    __shared__ float red_q[32];
    __shared__ float wo_lds[H * 4];
    __shared__ float bo_lds[4];

    const int  tid    = threadIdx.x;
    const int  lane   = tid & 31;
    const int  wid    = tid >> 5;          // 0..31
    const bool hiHalf = (lane >= 16);
    const int  ln     = lane & 15;

    // zero initial hidden state (each thread owns row `tid`)
    {
        float4 z4 = {0.f, 0.f, 0.f, 0.f};
#pragma unroll
        for (int j = 0; j < 8; ++j)
            *(float4*)&h_lds[tid * H + j * 4] = z4;
    }
    __syncthreads();

    const int kb4 = hiHalf ? 2 : 0;        // f32 K=4 A/B fragment K base

    for (int l = 0; l < L_LAYERS; ++l) {
        const float* Ur = Ur_w + l * H * H;
        const float* Uz = Uz_w + l * H * H;
        const float* Uh = Uh_w + l * H * H;
        const float* Wr = Wr_w + l * IN_DIM * H;
        const float* Wz = Wz_w + l * IN_DIM * H;
        const float* Wh = Wh_w + l * IN_DIM * H;

        // ---- stage wave-invariant B fragments into LDS -------------------
        // f16 B (K x N = 32x16): lane = column N (ln), lanes>=16 -> K 8..15,24..31,
        // element e<8 -> K 8h+e, e>=8 -> K 16+8h+(e-8).
        for (int f = tid; f < 3 * 2 * 32 * 16; f += 1024) {
            const int gate = f >> 10;             // / 1024
            const int rem  = f & 1023;
            const int nt   = rem >> 9;            // / 512
            const int lf   = (rem >> 4) & 31;     // lane
            const int e    = rem & 15;
            const int col  = nt * 16 + (lf & 15);
            const int kb   = (lf & 16) ? 8 : 0;
            const float* Up = (gate == 0) ? Ur : (gate == 1) ? Uz : Uh;
            ufrag[gate][nt][lf * UFRAG_STRIDE + e] =
                (_Float16)Up[(kb + e + (e & 8)) * H + col];
        }
        // f32 B (K x N = 4x16): lane = column N, lanes>=16 -> K 2..3.
        if (tid < 3 * 2 * 32 * 2) {
            const int gate = tid / 128;
            const int rem  = tid % 128;
            const int nt   = rem >> 6;
            const int lf   = (rem >> 1) & 31;
            const int e    = tid & 1;
            const int col  = nt * 16 + (lf & 15);
            const int kb   = (lf & 16) ? 2 : 0;
            const float* Wp = (gate == 0) ? Wr : (gate == 1) ? Wz : Wh;
            wfrag[gate][nt][lf * 2 + e] = Wp[(kb + e) * H + col];
        }

        float biasR[2], biasZ[2], biasHU[2], biasHX[2];
#pragma unroll
        for (int nt = 0; nt < 2; ++nt) {
            const int col = nt * 16 + ln;
            biasR[nt]  = Wr_b[l * H + col] + Ur_b[l * H + col];
            biasZ[nt]  = Wz_b[l * H + col] + Uz_b[l * H + col];
            biasHU[nt] = Uh_b[l * H + col];
            biasHX[nt] = Wh_b[l * H + col];
        }
        const float gbn  = bn_g[l];
        const float btn  = bn_b[l];
        const bool  last = (l == L_LAYERS - 1);
        if (last) {
            if (tid < H * 4) wo_lds[tid] = out_w[l * H * 4 + tid];
            if (tid < 4)     bo_lds[tid] = out_b[l * 4 + tid];
        }
        __syncthreads();

        const int tb0  = wid * 32;          // first M tile base row
        const int klo  = hiHalf ? 8 : 0;    // f16 A fragment K base
        const _Float16* uf = &ufrag[0][0][lane * UFRAG_STRIDE];
        const float*    wf = &wfrag[0][0][lane * 2];
        const int ufragNT = 32 * UFRAG_STRIDE;   // halfs per nt slot
        const int ufragG  = 2 * ufragNT;         // halfs per gate slot

        for (int t = 0; t < S_SEQ; ++t) {
            float preg[32];                 // candidate pre-activation (f32)
            v2h   zp[16];                   // update gate, packed f16 pairs
            float s_acc = 0.f, q_acc = 0.f;

            // ---------------- phase 1: gate matmuls via WMMA ----------------
#pragma unroll
            for (int mt = 0; mt < 2; ++mt) {
                const int tb  = tb0 + mt * 16;
                const int row = tb + ln;

                // A operand h-tile [16 x 32] f16 from f32 LDS master
                const float* hr = &h_lds[row * H];
                float4 f0 = *(const float4*)(hr + klo);
                float4 f1 = *(const float4*)(hr + klo + 4);
                float4 f2 = *(const float4*)(hr + klo + 16);
                float4 f3 = *(const float4*)(hr + klo + 20);
                v16h aH;
                aH[0]=(_Float16)f0.x;  aH[1]=(_Float16)f0.y;
                aH[2]=(_Float16)f0.z;  aH[3]=(_Float16)f0.w;
                aH[4]=(_Float16)f1.x;  aH[5]=(_Float16)f1.y;
                aH[6]=(_Float16)f1.z;  aH[7]=(_Float16)f1.w;
                aH[8]=(_Float16)f2.x;  aH[9]=(_Float16)f2.y;
                aH[10]=(_Float16)f2.z; aH[11]=(_Float16)f2.w;
                aH[12]=(_Float16)f3.x; aH[13]=(_Float16)f3.y;
                aH[14]=(_Float16)f3.z; aH[15]=(_Float16)f3.w;

                // A operand x-tile [16 x 4] f32 (global, tiny)
                const float2 x2 = *(const float2*)(in_seq +
                        ((size_t)row * S_SEQ + t) * IN_DIM + kb4);
                v2f aX; aX[0] = x2.x; aX[1] = x2.y;

#pragma unroll
                for (int nt = 0; nt < 2; ++nt) {
                    const int base = nt * ufragNT;

                    // ---- r gate ----
                    v8h lo = *(const v8h*)(uf + base);
                    v8h hi = *(const v8h*)(uf + base + 8);
                    v16h bu;
#pragma unroll
                    for (int i = 0; i < 8; ++i) { bu[i] = lo[i]; bu[8+i] = hi[i]; }
                    v2f bw = *(const v2f*)(wf + nt * 64);
                    v8f c = splat8(biasR[nt]);
                    c = __builtin_amdgcn_wmma_f32_16x16x32_f16(
                            false, aH, false, bu, (short)0, c, false, false);
                    c = __builtin_amdgcn_wmma_f32_16x16x4_f32(
                            false, aX, false, bw, (short)0, c, false, false);
                    float r8[8];
#pragma unroll
                    for (int i = 0; i < 8; ++i) r8[i] = sigmoidf_(c[i]);

                    // ---- z gate (store packed f16) ----
                    lo = *(const v8h*)(uf + ufragG + base);
                    hi = *(const v8h*)(uf + ufragG + base + 8);
#pragma unroll
                    for (int i = 0; i < 8; ++i) { bu[i] = lo[i]; bu[8+i] = hi[i]; }
                    bw = *(const v2f*)(wf + 2 * 128 + nt * 64);
                    c = splat8(biasZ[nt]);
                    c = __builtin_amdgcn_wmma_f32_16x16x32_f16(
                            false, aH, false, bu, (short)0, c, false, false);
                    c = __builtin_amdgcn_wmma_f32_16x16x4_f32(
                            false, aX, false, bw, (short)0, c, false, false);
#pragma unroll
                    for (int i2 = 0; i2 < 4; ++i2) {
                        v2h zz;
                        zz[0] = (_Float16)sigmoidf_(c[2 * i2]);
                        zz[1] = (_Float16)sigmoidf_(c[2 * i2 + 1]);
                        zp[mt * 8 + nt * 4 + i2] = zz;
                    }

                    // ---- candidate gate: pre = xWh+bh + r*(hUh+buh) ----
                    lo = *(const v8h*)(uf + 2 * ufragG + base);
                    hi = *(const v8h*)(uf + 2 * ufragG + base + 8);
#pragma unroll
                    for (int i = 0; i < 8; ++i) { bu[i] = lo[i]; bu[8+i] = hi[i]; }
                    bw = *(const v2f*)(wf + 4 * 128 + nt * 64);
                    v8f cu = splat8(biasHU[nt]);
                    cu = __builtin_amdgcn_wmma_f32_16x16x32_f16(
                            false, aH, false, bu, (short)0, cu, false, false);
                    v8f cx = splat8(biasHX[nt]);
                    cx = __builtin_amdgcn_wmma_f32_16x16x4_f32(
                            false, aX, false, bw, (short)0, cx, false, false);
#pragma unroll
                    for (int i = 0; i < 8; ++i) {
                        float p = fmaf(r8[i], cu[i], cx[i]);
                        preg[mt * 16 + nt * 8 + i] = p;
                        s_acc += p;
                        q_acc  = fmaf(p, p, q_acc);
                    }
                }
            }

            // -------- phase 2: batch-global mean/var (BatchNorm1d(1)) -------
#pragma unroll
            for (int off = 16; off > 0; off >>= 1) {
                s_acc += __shfl_xor(s_acc, off);
                q_acc += __shfl_xor(q_acc, off);
            }
            if (lane == 0) { red_s[wid] = s_acc; red_q[wid] = q_acc; }
            __syncthreads();
            // every wave reduces the 32 partials itself (no 2nd barrier)
            float ts = red_s[lane];
            float tq = red_q[lane];
#pragma unroll
            for (int off = 16; off > 0; off >>= 1) {
                ts += __shfl_xor(ts, off);
                tq += __shfl_xor(tq, off);
            }
            const float inv  = 1.0f / (float)(B_BATCH * H);
            const float mean = ts * inv;
            const float var  = fmaf(tq, inv, -mean * mean);
            const float sc   = gbn * __builtin_amdgcn_rsqf(var + EPS_BN);
            const float sh   = fmaf(-mean, sc, btn);

            // -------- phase 3: relu-norm + GRU update, write h back ---------
#pragma unroll
            for (int mt = 0; mt < 2; ++mt) {
                const int rowb = tb0 + mt * 16 + (hiHalf ? 8 : 0);
#pragma unroll
                for (int nt = 0; nt < 2; ++nt) {
                    const int hid = nt * 16 + ln;
#pragma unroll
                    for (int i2 = 0; i2 < 4; ++i2) {
                        const int idx = mt * 16 + nt * 8 + 2 * i2;
                        const v2h zz  = zp[mt * 8 + nt * 4 + i2];

                        float prop0 = fmaf(preg[idx], sc, sh);
                        prop0 = fmaxf(prop0, 0.0f);
                        const int off0 = (rowb + 2 * i2) * H + hid;
                        float hc0 = h_lds[off0];
                        h_lds[off0] = fmaf((float)zz[0], prop0 - hc0, hc0);

                        float prop1 = fmaf(preg[idx + 1], sc, sh);
                        prop1 = fmaxf(prop1, 0.0f);
                        const int off1 = off0 + H;
                        float hc1 = h_lds[off1];
                        h_lds[off1] = fmaf((float)zz[1], prop1 - hc1, hc1);
                    }
                }
            }
            __syncthreads();

            // -------- last layer: out = hn @ Wo + bo  (row = tid) -----------
            if (last) {
                const float* hr = &h_lds[tid * H];
                float a0 = bo_lds[0], a1 = bo_lds[1];
                float a2 = bo_lds[2], a3 = bo_lds[3];
#pragma unroll
                for (int k = 0; k < H; ++k) {
                    float hv = hr[k];
                    a0 = fmaf(hv, wo_lds[k * 4 + 0], a0);
                    a1 = fmaf(hv, wo_lds[k * 4 + 1], a1);
                    a2 = fmaf(hv, wo_lds[k * 4 + 2], a2);
                    a3 = fmaf(hv, wo_lds[k * 4 + 3], a3);
                }
                float4 o = {a0, a1, a2, a3};
                *(float4*)(outs + ((size_t)tid * S_SEQ + t) * 4) = o;
            }
        } // t
    } // l
}

// ---------------------------------------------------------------------------
// Physics pass: denormalize + fixed-step dopri5, fully parallel over B*S.
// d_out layout: [Tair_f (BS)] [out (BS*3)] [Tair_f (BS)]
// ---------------------------------------------------------------------------
__global__ void physics_kernel(const float* __restrict__ outs,
                               const float* __restrict__ max_v,
                               const float* __restrict__ min_v,
                               float* __restrict__ dout)
{
    const int N   = B_BATCH * S_SEQ;
    const int idx = blockIdx.x * blockDim.x + threadIdx.x;
    if (idx >= N) return;

    float4 o = *(const float4*)(outs + (size_t)idx * 4);
    const float Irad = o.x * (max_v[0] - min_v[0]) + min_v[0];
    const float Tout = o.y * (max_v[1] - min_v[1]) + min_v[1];
    const float Hair = o.z * (max_v[2] - min_v[2]) + min_v[2];
    const float Tair = o.w * (max_v[3] - min_v[3]) + min_v[3];

    const float Qsun   = 10.0f * 0.7f * Irad;
    const float Rn     = 0.86f * (1.0f - __expf(-0.7f * 2.6f)) * Qsun;
    const float rsBase = 82.0f + 570.0f * __expf(-(1e-4f / 2.6f) * Rn);
    const float HcropC = (150.0f / (2.0f * 2.6f)) * (Rn / 2450.0f);
    const float hs     = 60.0f / (float)ODE_N;

    auto dydt = [&](float Ta) -> float {
        Ta = fminf(Ta, 500.0f);
        float Qcov   = 5.0f * (Ta - Tout);
        float Hsat   = 5.5638f * __expf(0.0572f * Ta);
        float ce     = 0.7584f * __expf(0.0518f * Ta);
        float d      = Ta - 5.0f;
        float rs     = rsBase * (1.0f + 0.023f * d * d);
        float ge     = (2.0f * 2.6f) *
                       __builtin_amdgcn_rcpf((1.0f + ce) * 150.0f + rs);
        float Hcrop  = Hsat + ce * HcropC;
        float Qtrans = ge * (-2450.0f) * (Hcrop - Hair);
        float Qvent  = -(0.3f * 1.225f * 1005.0f) * (Ta - Tout);
        return (-1.0f / 1000.0f) * (Qsun - Qcov - Qtrans - Qvent);
    };

    float y = Tair;
    for (int s = 0; s < ODE_N; ++s) {
        float k1 = dydt(y);
        float k2 = dydt(y + hs * (0.2f * k1));
        float k3 = dydt(y + hs * (0.075f * k1 + 0.225f * k2));
        float k4 = dydt(y + hs * ((44.0f/45.0f)*k1 - (56.0f/15.0f)*k2
                                  + (32.0f/9.0f)*k3));
        float k5 = dydt(y + hs * ((19372.0f/6561.0f)*k1 - (25360.0f/2187.0f)*k2
                                  + (64448.0f/6561.0f)*k3 - (212.0f/729.0f)*k4));
        float k6 = dydt(y + hs * ((9017.0f/3168.0f)*k1 - (355.0f/33.0f)*k2
                                  + (46732.0f/5247.0f)*k3 + (49.0f/176.0f)*k4
                                  - (5103.0f/18656.0f)*k5));
        y = y + hs * ((35.0f/384.0f)*k1 + (500.0f/1113.0f)*k3
                      + (125.0f/192.0f)*k4 - (2187.0f/6784.0f)*k5
                      + (11.0f/84.0f)*k6);
    }

    dout[idx]         = y;           // Tair_f (first return)
    dout[4 * N + idx] = y;           // Tair_f (third return)
    float* o3 = dout + N + (size_t)idx * 3;
    o3[0] = Irad; o3[1] = Tout; o3[2] = Hair;
}

// ---------------------------------------------------------------------------
extern "C" void kernel_launch(void* const* d_in, const int* in_sizes, int n_in,
                              void* d_out, int out_size, void* d_ws, size_t ws_size,
                              hipStream_t stream)
{
    (void)in_sizes; (void)n_in; (void)out_size; (void)ws_size;

    const float* in_seq = (const float*)d_in[0];
    const float* max_v  = (const float*)d_in[1];
    const float* min_v  = (const float*)d_in[2];
    const float* Wr_w   = (const float*)d_in[3];
    const float* Wr_b   = (const float*)d_in[4];
    const float* Ur_w   = (const float*)d_in[5];
    const float* Ur_b   = (const float*)d_in[6];
    const float* Wz_w   = (const float*)d_in[7];
    const float* Wz_b   = (const float*)d_in[8];
    const float* Uz_w   = (const float*)d_in[9];
    const float* Uz_b   = (const float*)d_in[10];
    const float* Wh_w   = (const float*)d_in[11];
    const float* Wh_b   = (const float*)d_in[12];
    const float* Uh_w   = (const float*)d_in[13];
    const float* Uh_b   = (const float*)d_in[14];
    const float* out_w  = (const float*)d_in[15];
    const float* out_b  = (const float*)d_in[16];
    const float* bn_g   = (const float*)d_in[17];
    const float* bn_b   = (const float*)d_in[18];

    float* outs_ws = (float*)d_ws;   // [B, S, 4] f32 = 8 MB scratch

    gru_persistent<<<1, 1024, 0, stream>>>(
        in_seq, Wr_w, Wr_b, Ur_w, Ur_b, Wz_w, Wz_b, Uz_w, Uz_b,
        Wh_w, Wh_b, Uh_w, Uh_b, out_w, out_b, bn_g, bn_b, outs_ws);

    const int N = B_BATCH * S_SEQ;
    physics_kernel<<<(N + 255) / 256, 256, 0, stream>>>(
        outs_ws, max_v, min_v, (float*)d_out);
}
